// GraphActionNetwork_18322330485509
// MI455X (gfx1250) — matrix operand order repaired and checked by reading, not compiled, not run on hardware
//
#include <hip/hip_runtime.h>

typedef unsigned int   u32;
typedef unsigned short u16;
typedef __attribute__((ext_vector_type(16))) __bf16 v16bf;
typedef __attribute__((ext_vector_type(8)))  float  v8f;
typedef __attribute__((ext_vector_type(4)))  u32    v4u;

#define N_NODES 8192
#define EMBED   64
#define BATCH   4096
#define H1      512
#define H2      256
#define H3      128
#define ACT     8
#define INSZ    (2*N_NODES + 2*EMBED)   /* 16512 */

union Frag { v16bf bf; v4u u[2]; };

// RNE f32->bf16 (used in one-time conversion passes, off the GEMM critical path)
__device__ __forceinline__ u32 f2bf(float f) {
  u32 u = __builtin_bit_cast(u32, f);
  return ((u + 0x7FFFu + ((u >> 16) & 1u)) >> 16) & 0xFFFFu;
}

// single-instruction truncating pack of two f32 into {bf16(b),bf16(a)} via v_perm_b32
__device__ __forceinline__ u32 pk2t(float a, float b) {
  return __builtin_amdgcn_perm(__builtin_bit_cast(u32, b),
                               __builtin_bit_cast(u32, a), 0x07060302u);
}
__device__ __forceinline__ v4u pack8t(const float* __restrict__ p) {
  const float4* q = (const float4*)p;
  float4 x = q[0], y = q[1];
  v4u r;
  r.x = pk2t(x.x, x.y); r.y = pk2t(x.z, x.w);
  r.z = pk2t(y.x, y.y); r.w = pk2t(y.z, y.w);
  return r;
}

// build one WMMA A fragment (16 bf16) for this lane
template <bool ABF16>
__device__ __forceinline__ Frag loadA(const void* __restrict__ Arow, int kbase) {
  Frag a;
  if (ABF16) {
    const u16* r = (const u16*)Arow;
    a.u[0] = *(const v4u*)(r + kbase);
    a.u[1] = *(const v4u*)(r + kbase + 16);
  } else {
    const float* r = (const float*)Arow;
    a.u[0] = pack8t(r + kbase);
    a.u[1] = pack8t(r + kbase + 16);
  }
  return a;
}

// ---------------- diag extract ----------------
__global__ void k_diag(const float* __restrict__ state, float* __restrict__ diag) {
  int i = blockIdx.x * blockDim.x + threadIdx.x;
  if (i < N_NODES) diag[i] = state[(size_t)i * N_NODES + i];
}

// ---- adjx[i] = sum_n state[i,n]*diag[n] - diag[i]^2 ; optional fused bf16 copy of state ----
__global__ void __launch_bounds__(256) k_adjx(const float* __restrict__ state,
                                              const float* __restrict__ diag,
                                              float* __restrict__ adjx,
                                              u16* __restrict__ state_bf) {
  __shared__ float red[256];
  int row = blockIdx.x, t = threadIdx.x;
  const float* r = state + (size_t)row * N_NODES;
  u16* wb = state_bf ? state_bf + (size_t)row * N_NODES : nullptr;
  float s = 0.f;
  for (int n = t; n < N_NODES; n += 256) {
    float x = r[n];
    s = fmaf(x, diag[n], s);
    if (wb) wb[n] = (u16)f2bf(x);
  }
  red[t] = s; __syncthreads();
  for (int w = 128; w > 0; w >>= 1) { if (t < w) red[t] += red[t + w]; __syncthreads(); }
  if (t == 0) adjx[row] = red[0] - diag[row] * diag[row];
}

// ------- g[i,k] = sum_j relu(adjx[i]*gw1[j] + gb1[j]) * gw2[j,k]   (h1 never materialized) -------
__global__ void __launch_bounds__(256) k_g(const float* __restrict__ adjx,
                                           const float* __restrict__ gw1,
                                           const float* __restrict__ gb1,
                                           const float* __restrict__ gw2,
                                           float* __restrict__ g) {
  __shared__ float sw[H1], sb[H1];
  int t = threadIdx.x;
  for (int j = t; j < H1; j += 256) { sw[j] = gw1[j]; sb[j] = gb1[j]; }
  __syncthreads();
  int k = t & 63, r = (t >> 6);
  int row = blockIdx.x * 4 + r;
  float a = adjx[row], acc = 0.f;
  for (int j = 0; j < H1; ++j) {
    float hv = fmaxf(fmaf(a, sw[j], sb[j]), 0.f);
    acc = fmaf(hv, gw2[j * EMBED + k], acc);
  }
  g[(size_t)row * EMBED + k] = acc;
}

// ------- swizzle f32 weight [K,N] into wave32 WMMA-bf16 B-fragment layout -------
__global__ void k_swz(const float* __restrict__ W, u16* __restrict__ out,
                      int N, int ncChunks, int total) {
  int o = blockIdx.x * blockDim.x + threadIdx.x;
  if (o >= total) return;
  int e  = o & 15;
  int L  = (o >> 4) & 31;
  int ch = o >> 9;
  int nc = ch % ncChunks, kc = ch / ncChunks;
  int k = kc * 32 + ((L < 16) ? 0 : 8) + (e & 7) + ((e < 8) ? 0 : 16);
  int n = nc * 16 + (L & 15);
  out[o] = (u16)f2bf(W[(size_t)k * N + n]);
}

// ------- generic WMMA GEMM: C[M,N] = (relu?)(A[M,K] @ Bswz + bias), optional bf16 C copy -------
template <bool ABF16>
__global__ void __launch_bounds__(256) k_gemm(const void* __restrict__ A, int lda,
                                              const u16* __restrict__ Bswz,
                                              const float* __restrict__ bias,
                                              float* __restrict__ C,
                                              u16* __restrict__ Cbf, int ldc,
                                              int N, int K, int relu) {
  int lane = threadIdx.x & 31, wave = threadIdx.x >> 5;
  int m0 = (blockIdx.x * 8 + wave) * 16;
  int n0 = blockIdx.y * 64;
  int nchunks = N >> 4, nc0 = n0 >> 4;
  int row = m0 + (lane & 15);
  int kb  = (lane < 16) ? 0 : 8;
  const void* arow = ABF16 ? (const void*)((const u16*)A + (size_t)row * lda)
                           : (const void*)((const float*)A + (size_t)row * lda);
  v8f z = {0.f,0.f,0.f,0.f,0.f,0.f,0.f,0.f};
  v8f acc[4] = { z, z, z, z };
  for (int kc = 0; kc < (K >> 5); ++kc) {
    int kbase = (kc << 5) + kb;
    if (ABF16) __builtin_prefetch((const u16*)arow + kbase + 128, 0, 3);
    else       __builtin_prefetch((const float*)arow + kbase + 64, 0, 3);
    // issue ALL loads for this k-step first, then the 4 WMMAs (partial waits overlap)
    Frag a = loadA<ABF16>(arow, kbase);
    Frag b[4];
    const u16* bb = Bswz + (((size_t)kc * nchunks + nc0) * 32 + lane) * 16;
#pragma unroll
    for (int f = 0; f < 4; ++f) {
      const v4u* bp = (const v4u*)(bb + (size_t)f * 512);
      b[f].u[0] = bp[0]; b[f].u[1] = bp[1];
    }
#pragma unroll
    for (int f = 0; f < 4; ++f)
      acc[f] = __builtin_amdgcn_wmma_f32_16x16x32_bf16(false, a.bf, false, b[f].bf,
                                                       (short)0, acc[f], false, false);
  }
  int rowoff = m0 + ((lane < 16) ? 0 : 8);
#pragma unroll
  for (int f = 0; f < 4; ++f) {
    int col = n0 + f * 16 + (lane & 15);
    float bv = bias ? bias[col] : 0.f;
#pragma unroll
    for (int r = 0; r < 8; ++r) {
      float v = acc[f][r] + bv;
      if (relu) v = fmaxf(v, 0.f);
      size_t idx = (size_t)(rowoff + r) * ldc + col;
      C[idx] = v;
      if (Cbf) Cbf[idx] = (u16)f2bf(v);
    }
  }
}

// ------- diag correction + gb2 + relu + row log_softmax -> h [8192,64] (+bf16 copy) -------
__global__ void __launch_bounds__(64) k_fix_softmax(const float* __restrict__ s,
                                                    const float* __restrict__ g,
                                                    const float* __restrict__ diag,
                                                    const float* __restrict__ gb2,
                                                    float* __restrict__ h,
                                                    u16* __restrict__ hbf) {
  __shared__ float red[64];
  int row = blockIdx.x, t = threadIdx.x;
  size_t idx = (size_t)row * EMBED + t;
  float v = fmaxf(s[idx] - diag[row] * g[idx] + gb2[t], 0.f);
  red[t] = v; __syncthreads();
  for (int w = 32; w > 0; w >>= 1) { if (t < w) red[t] = fmaxf(red[t], red[t + w]); __syncthreads(); }
  float mx = red[0]; __syncthreads();
  red[t] = expf(v - mx); __syncthreads();
  for (int w = 32; w > 0; w >>= 1) { if (t < w) red[t] += red[t + w]; __syncthreads(); }
  float o = v - mx - logf(red[0]);
  h[idx] = o;
  if (hbf) hbf[idx] = (u16)f2bf(o);
}

// ------- head layer 1: virtual gathered A = [state[start]|state[end]|h[start]|h[end]] @ fw1 -------
template <bool ABF16>
__global__ void __launch_bounds__(256) k_head(const void* __restrict__ stateA,
                                              const void* __restrict__ hA,
                                              const int* __restrict__ start,
                                              const int* __restrict__ endi,
                                              const u16* __restrict__ fw1swz,
                                              const float* __restrict__ fb1,
                                              float* __restrict__ y1,
                                              u16* __restrict__ y1bf) {
  int lane = threadIdx.x & 31, wave = threadIdx.x >> 5;
  int m0 = (blockIdx.x * 8 + wave) * 16;
  int n0 = blockIdx.y * 64;
  int b  = m0 + (lane & 15);
  int es = ABF16 ? 2 : 4;  // element size in bytes
  const char* pS  = (const char*)stateA + (size_t)start[b] * N_NODES * es;
  const char* pE  = (const char*)stateA + (size_t)endi[b]  * N_NODES * es;
  const char* pHs = (const char*)hA + (size_t)start[b] * EMBED * es;
  const char* pHe = (const char*)hA + (size_t)endi[b]  * EMBED * es;
  int kb = (lane < 16) ? 0 : 8;
  int nchunks = H1 >> 4, nc0 = n0 >> 4;
  v8f z = {0.f,0.f,0.f,0.f,0.f,0.f,0.f,0.f};
  v8f acc[4] = { z, z, z, z };
  for (int kc = 0; kc < (INSZ >> 5); ++kc) {
    int k = kc << 5;
    const char* src; int off;
    if      (k < N_NODES)     { src = pS;  off = k; }
    else if (k < 2 * N_NODES) { src = pE;  off = k - N_NODES; }
    else if (k < 2 * N_NODES + EMBED) { src = pHs; off = k - 2 * N_NODES; }
    else                      { src = pHe; off = k - 2 * N_NODES - EMBED; }
    __builtin_prefetch(src + (size_t)(off + 32) * es, 0, 3);
    Frag a = loadA<ABF16>(src + (size_t)off * es, kb);
    Frag bm[4];
    const u16* bb = fw1swz + (((size_t)kc * nchunks + nc0) * 32 + lane) * 16;
#pragma unroll
    for (int f = 0; f < 4; ++f) {
      const v4u* bp = (const v4u*)(bb + (size_t)f * 512);
      bm[f].u[0] = bp[0]; bm[f].u[1] = bp[1];
    }
#pragma unroll
    for (int f = 0; f < 4; ++f)
      acc[f] = __builtin_amdgcn_wmma_f32_16x16x32_bf16(false, a.bf, false, bm[f].bf,
                                                       (short)0, acc[f], false, false);
  }
  int rowoff = m0 + ((lane < 16) ? 0 : 8);
#pragma unroll
  for (int f = 0; f < 4; ++f) {
    int col = n0 + f * 16 + (lane & 15);
    float bv = fb1[col];
#pragma unroll
    for (int r = 0; r < 8; ++r) {
      float v = fmaxf(acc[f][r] + bv, 0.f);
      size_t idx = (size_t)(rowoff + r) * H1 + col;
      y1[idx] = v;
      if (y1bf) y1bf[idx] = (u16)f2bf(v);
    }
  }
}

// ------- final tiny layer: q = relu(y3 @ fw4 + fb4) -------
__global__ void k_last(const float* __restrict__ y3, const float* __restrict__ fw4,
                       const float* __restrict__ fb4, float* __restrict__ out) {
  int tid = blockIdx.x * blockDim.x + threadIdx.x;
  if (tid >= BATCH * ACT) return;
  int b = tid >> 3, a = tid & 7;
  float s = fb4[a];
  const float* r = y3 + (size_t)b * H3;
  for (int j = 0; j < H3; ++j) s = fmaf(r[j], fw4[j * ACT + a], s);
  out[tid] = fmaxf(s, 0.f);
}

extern "C" void kernel_launch(void* const* d_in, const int* in_sizes, int n_in,
                              void* d_out, int out_size, void* d_ws, size_t ws_size,
                              hipStream_t stream) {
  (void)in_sizes; (void)n_in; (void)out_size;
  const float* state = (const float*)d_in[0];
  const int*   start = (const int*)  d_in[1];
  const int*   endi  = (const int*)  d_in[2];
  const float* gw1   = (const float*)d_in[3];
  const float* gb1   = (const float*)d_in[4];
  const float* gw2   = (const float*)d_in[5];
  const float* gb2   = (const float*)d_in[6];
  const float* fw1   = (const float*)d_in[7];
  const float* fb1   = (const float*)d_in[8];
  const float* fw2   = (const float*)d_in[9];
  const float* fb2   = (const float*)d_in[10];
  const float* fw3   = (const float*)d_in[11];
  const float* fb3   = (const float*)d_in[12];
  const float* fw4   = (const float*)d_in[13];
  const float* fb4   = (const float*)d_in[14];
  float* out = (float*)d_out;

  char* p = (char*)d_ws;
  size_t o = 0;
  auto alloc = [&](size_t bytes) { void* r = p + o; o = (o + bytes + 255) & ~(size_t)255; return r; };
  float* diag   = (float*)alloc(N_NODES * 4);
  float* adjx   = (float*)alloc(N_NODES * 4);
  float* g      = (float*)alloc((size_t)N_NODES * EMBED * 4);
  float* s      = (float*)alloc((size_t)N_NODES * EMBED * 4);
  float* h      = (float*)alloc((size_t)N_NODES * EMBED * 4);
  float* y1     = (float*)alloc((size_t)BATCH * H1 * 4);
  float* y2     = (float*)alloc((size_t)BATCH * H2 * 4);
  float* y3     = (float*)alloc((size_t)BATCH * H3 * 4);
  u16*   gswz   = (u16*)alloc((size_t)N_NODES * EMBED * 2);
  u16*   fw1swz = (u16*)alloc((size_t)INSZ * H1 * 2);
  u16*   fw2swz = (u16*)alloc((size_t)H1 * H2 * 2);
  u16*   fw3swz = (u16*)alloc((size_t)H2 * H3 * 2);
  size_t common = o;
  size_t extras = ((size_t)N_NODES * N_NODES * 2 + 256)   // state_bf
                + ((size_t)N_NODES * EMBED * 2 + 256)     // h_bf
                + ((size_t)BATCH * H1 * 2 + 256)          // y1_bf
                + ((size_t)BATCH * H2 * 2 + 256);         // y2_bf
  bool big = (ws_size >= common + extras);
  u16 *state_bf = nullptr, *h_bf = nullptr, *y1_bf = nullptr, *y2_bf = nullptr;
  if (big) {
    state_bf = (u16*)alloc((size_t)N_NODES * N_NODES * 2);
    h_bf     = (u16*)alloc((size_t)N_NODES * EMBED * 2);
    y1_bf    = (u16*)alloc((size_t)BATCH * H1 * 2);
    y2_bf    = (u16*)alloc((size_t)BATCH * H2 * 2);
  }

  // 1. diag + rank-1-collapsed GCN layer 1 (matvec pass doubles as f32->bf16 conversion of state)
  k_diag<<<32, 256, 0, stream>>>(state, diag);
  k_adjx<<<N_NODES, 256, 0, stream>>>(state, diag, adjx, state_bf);
  k_g<<<N_NODES / 4, 256, 0, stream>>>(adjx, gw1, gb1, gw2, g);

  // 2. weight/activation swizzles into WMMA B layout
  { int t = N_NODES * EMBED; k_swz<<<(t + 255) / 256, 256, 0, stream>>>(g,   gswz,   EMBED, EMBED / 16, t); }
  { int t = INSZ * H1;       k_swz<<<(t + 255) / 256, 256, 0, stream>>>(fw1, fw1swz, H1, H1 / 16, t); }
  { int t = H1 * H2;         k_swz<<<(t + 255) / 256, 256, 0, stream>>>(fw2, fw2swz, H2, H2 / 16, t); }
  { int t = H2 * H3;         k_swz<<<(t + 255) / 256, 256, 0, stream>>>(fw3, fw3swz, H3, H3 / 16, t); }

  // 3. GCN layer 2: s = state @ g (WMMA), then diag-fix + bias + relu + log_softmax
  if (big)
    k_gemm<true><<<dim3(N_NODES / 128, EMBED / 64), 256, 0, stream>>>(
        state_bf, N_NODES, gswz, nullptr, s, nullptr, EMBED, EMBED, N_NODES, 0);
  else
    k_gemm<false><<<dim3(N_NODES / 128, EMBED / 64), 256, 0, stream>>>(
        state, N_NODES, gswz, nullptr, s, nullptr, EMBED, EMBED, N_NODES, 0);
  k_fix_softmax<<<N_NODES, 64, 0, stream>>>(s, g, diag, gb2, h, h_bf);

  // 4. head layer 1 (gathered A, WMMA), layers 2-3 (WMMA), layer 4 (VALU)
  if (big) {
    k_head<true><<<dim3(BATCH / 128, H1 / 64), 256, 0, stream>>>(
        state_bf, h_bf, start, endi, fw1swz, fb1, y1, y1_bf);
    k_gemm<true><<<dim3(BATCH / 128, H2 / 64), 256, 0, stream>>>(
        y1_bf, H1, fw2swz, fb2, y2, y2_bf, H2, H2, H1, 1);
    k_gemm<true><<<dim3(BATCH / 128, H3 / 64), 256, 0, stream>>>(
        y2_bf, H2, fw3swz, fb3, y3, nullptr, H3, H3, H2, 1);
  } else {
    k_head<false><<<dim3(BATCH / 128, H1 / 64), 256, 0, stream>>>(
        state, h, start, endi, fw1swz, fb1, y1, nullptr);
    k_gemm<false><<<dim3(BATCH / 128, H2 / 64), 256, 0, stream>>>(
        y1, H1, fw2swz, fb2, y2, nullptr, H2, H2, H1, 1);
    k_gemm<false><<<dim3(BATCH / 128, H3 / 64), 256, 0, stream>>>(
        y2, H2, fw3swz, fb3, y3, nullptr, H3, H3, H2, 1);
  }
  k_last<<<(BATCH * ACT + 255) / 256, 256, 0, stream>>>(y3, fw4, fb4, out);
}
